// Decoder_30227979829552
// MI455X (gfx1250) — matrix operand order repaired and checked by reading, compiled, not run on hardware
//
#include <hip/hip_runtime.h>
#include <hip/hip_bf16.h>
#include <math.h>

// Optional gfx1250 async-to-LDS path (guarded; falls back to plain LDS copy)
#if defined(__has_builtin)
#if __has_builtin(__builtin_amdgcn_global_load_async_to_lds_b128) && \
    __has_builtin(__builtin_amdgcn_s_wait_asynccnt)
#define USE_ASYNC_LDS 1
#endif
#endif

namespace {
constexpr int Bn  = 128;
constexpr int Ln  = 512;
constexpr int En  = 128;
constexpr int Hn  = 256;
constexpr int KXH = En + Hn;   // 384 : [x ; h] concat width
constexpr int G4H = 4 * Hn;    // 1024: gate width
constexpr int H2  = 2 * Hn;    // 512 : [hs ; h_t] concat width

typedef __attribute__((ext_vector_type(16))) __bf16         v16bf;
typedef __attribute__((ext_vector_type(16))) unsigned short v16us;
typedef __attribute__((ext_vector_type(8)))  float          v8f;
typedef __attribute__((ext_vector_type(8)))  unsigned int   v8u;
typedef __attribute__((ext_vector_type(4)))  unsigned int   v4u;
typedef __attribute__((ext_vector_type(4)))  int            i4;
typedef __attribute__((address_space(1)))    i4             gi4;  // global int4
typedef __attribute__((address_space(3)))    i4             li4;  // LDS int4

__device__ __forceinline__ unsigned short f2bf(float f) {
  unsigned u = __builtin_bit_cast(unsigned, f);
  u += 0x7FFFu + ((u >> 16) & 1u);          // round-to-nearest-even
  return (unsigned short)(u >> 16);
}
__device__ __forceinline__ float bf2f(unsigned short h) {
  return __builtin_bit_cast(float, ((unsigned)h) << 16);
}
__device__ __forceinline__ float sigf(float x) { return 1.0f / (1.0f + __expf(-x)); }

// ---- WMMA fragment loaders (v_wmma_f32_16x16x32_bf16, wave32, ISA 7.12.2) ----

// A fragment: 16x32 (MxK) from row-major bf16 (global or LDS), two b128 loads/lane.
// lane<16: M=lane, K in {0..7,16..23}; lane>=16: M=lane-16, K in {8..15,24..31}.
__device__ __forceinline__ v16bf load_a_bf16(const unsigned short* A, int lda, int m0, int k0) {
  const int lane = threadIdx.x & 31;
  const unsigned short* row =
      A + (size_t)(m0 + (lane & 15)) * lda + k0 + ((lane >> 4) << 3);
  v4u lo = *(const v4u*)row;          // K = khi + 0..7
  v4u hi = *(const v4u*)(row + 16);   // K = khi + 16..23
  v8u d;
#pragma unroll
  for (int j = 0; j < 4; ++j) { d[j] = lo[j]; d[4 + j] = hi[j]; }
  return __builtin_bit_cast(v16bf, d);
}

// A fragment converted on the fly from f32 (one-time ctx projection only).
__device__ __forceinline__ v16bf load_a_f32(const float* A, int lda, int m0, int k0) {
  const int lane = threadIdx.x & 31;
  const float* row = A + (size_t)(m0 + (lane & 15)) * lda + k0 + ((lane >> 4) << 3);
  v16us r;
#pragma unroll
  for (int j = 0; j < 8; ++j) r[j] = f2bf(row[j]);
#pragma unroll
  for (int j = 0; j < 8; ++j) r[8 + j] = f2bf(row[16 + j]);
  return __builtin_bit_cast(v16bf, r);
}

// B fragment from pre-packed panels: weights are repacked once so each lane's
// 16 bf16 values are contiguous (32B) -> one v8u (two b128) load per fragment.
// Panel = 32K x 16N block, flat index = panel*512 + lane*16 + j.
__device__ __forceinline__ v16bf load_b_packed(const unsigned short* Bp, int N, int k0, int n0) {
  const int lane = threadIdx.x & 31;
  const size_t base =
      ((size_t)(k0 >> 5) * (N >> 4) + (n0 >> 4)) * 512 + (size_t)lane * 16;
  v8u d = *(const v8u*)(Bp + base);
  return __builtin_bit_cast(v16bf, d);
}

__device__ __forceinline__ v8f wmma_bf16(v16bf a, v16bf b, v8f c) {
  return __builtin_amdgcn_wmma_f32_16x16x32_bf16(false, a, false, b, (short)0, c,
                                                 false, false);
}

// ------------------------------- kernels -----------------------------------

// Repack f32 weight matrix [rows x N] (row offset k_off in the packed K dim)
// into WMMA B-fragment panel layout (bf16).
__global__ void pack_b_kernel(unsigned short* __restrict__ dst,
                              const float* __restrict__ src,
                              int rows, int N, int k_off) {
  int i = blockIdx.x * blockDim.x + threadIdx.x;
  if (i >= rows * N) return;
  const int k = k_off + i / N;
  const int n = i % N;
  const size_t fl = ((size_t)(k >> 5) * (N >> 4) + (n >> 4)) * 512 +
                    (size_t)((((k >> 3) & 1) << 4) + (n & 15)) * 16 + (k & 7) + (((k >> 4) & 1) << 3);
  dst[fl] = f2bf(src[i]);
}

// Per-call state init: xh = [bf16(decoder_input) ; bf16(h0)], c = c0, mask = 1
__global__ void init_state_kernel(const float* __restrict__ dec,
                                  const float* __restrict__ h0,
                                  const float* __restrict__ c0,
                                  unsigned short* __restrict__ xh,
                                  float* __restrict__ cstate,
                                  float* __restrict__ mask) {
  int i = blockIdx.x * blockDim.x + threadIdx.x;       // 0..Bn*Ln-1
  if (i < Bn * En) xh[(i / En) * KXH + (i % En)] = f2bf(dec[i]);
  if (i < Bn * Hn) {
    xh[(i / Hn) * KXH + En + (i % Hn)] = f2bf(h0[i]);
    cstate[i] = c0[i];
  }
  if (i < Bn * Ln) mask[i] = 1.0f;
}

// ctx = context @ W_ctx + b_ctx (M=65536, N=K=256). Writes f32 (for hs pass)
// and a bf16 mirror (for the per-step score pass: half the L2 traffic).
__global__ __launch_bounds__(256) void ctx_proj_kernel(const float* __restrict__ ctx_in,
                                                       const unsigned short* __restrict__ Wctx,
                                                       const float* __restrict__ b_ctx,
                                                       float* __restrict__ ctx_out,
                                                       unsigned short* __restrict__ ctx16) {
  const int wid = blockIdx.x * 8 + (threadIdx.x >> 5);
  const int m0  = (wid >> 4) * 16;
  const int n0  = (wid & 15) * 16;
  v8f acc = {};
#pragma unroll
  for (int k = 0; k < Hn; k += 32) {
    v16bf a = load_a_f32(ctx_in, Hn, m0, k);
    v16bf b = load_b_packed(Wctx, Hn, k, n0);
    acc = wmma_bf16(a, b, acc);
  }
  const int lane = threadIdx.x & 31;
  const int n    = n0 + (lane & 15);
  const int mb   = m0 + ((lane >> 4) << 3);
  const float bias = b_ctx[n];
#pragma unroll
  for (int v = 0; v < 8; ++v) {
    const float val = acc[v] + bias;
    ctx_out[(size_t)(mb + v) * Hn + n] = val;
    ctx16[(size_t)(mb + v) * Hn + n]   = f2bf(val);
  }
}

// gates = xh @ Wg + b; fused LSTM cell. A-panel (16x384 bf16, shared by the
// block's 8 waves) is staged to LDS -- via async-to-LDS when available.
__global__ __launch_bounds__(256) void lstm_gates_kernel(const unsigned short* __restrict__ xh,
                                                         const unsigned short* __restrict__ Wg,
                                                         const float* __restrict__ b_ih,
                                                         const float* __restrict__ b_hh,
                                                         float* __restrict__ cstate,
                                                         unsigned short* __restrict__ hsht) {
  __shared__ alignas(32) unsigned short sA[16 * KXH];   // 12 KB
  const int m0 = (blockIdx.x >> 1) * 16;                // one M-tile per block pair
  const unsigned short* gsrc = xh + (size_t)m0 * KXH;   // contiguous 16-row panel

#if defined(USE_ASYNC_LDS)
#pragma unroll
  for (int c = 0; c < 3; ++c) {                         // 768 x 16B chunks
    const int idx = threadIdx.x + c * 256;
    __builtin_amdgcn_global_load_async_to_lds_b128(
        (gi4*)(gsrc + idx * 8),
        (li4*)(sA + idx * 8), 0, 0);
  }
  __builtin_amdgcn_s_wait_asynccnt(0);
#else
#pragma unroll
  for (int c = 0; c < 3; ++c) {
    const int idx = threadIdx.x + c * 256;
    *(v4u*)(sA + idx * 8) = *(const v4u*)(gsrc + idx * 8);
  }
#endif
  __syncthreads();

  const int wid = blockIdx.x * 8 + (threadIdx.x >> 5);
  const int n0  = (wid & 15) * 16;
  v8f acc[4] = {v8f{}, v8f{}, v8f{}, v8f{}};
#pragma unroll
  for (int k = 0; k < KXH; k += 32) {
    v16bf a = load_a_bf16(sA, KXH, 0, k);               // ds_load path
#pragma unroll
    for (int g = 0; g < 4; ++g) {
      v16bf b = load_b_packed(Wg, G4H, k, g * Hn + n0);
      acc[g] = wmma_bf16(a, b, acc[g]);
    }
  }
  const int lane = threadIdx.x & 31;
  const int n    = n0 + (lane & 15);
  const int mb   = m0 + ((lane >> 4) << 3);
  const float bi = b_ih[n]          + b_hh[n];
  const float bf = b_ih[Hn + n]     + b_hh[Hn + n];
  const float bg = b_ih[2 * Hn + n] + b_hh[2 * Hn + n];
  const float bo = b_ih[3 * Hn + n] + b_hh[3 * Hn + n];
#pragma unroll
  for (int v = 0; v < 8; ++v) {
    const int m = mb + v;
    float gi = sigf(acc[0][v] + bi);
    float gf = sigf(acc[1][v] + bf);
    float gg = tanhf(acc[2][v] + bg);
    float go = sigf(acc[3][v] + bo);
    float ct = gf * cstate[m * Hn + n] + gi * gg;
    float ht = go * tanhf(ct);
    cstate[m * Hn + n]    = ct;
    hsht[m * H2 + Hn + n] = f2bf(ht);   // h_t into second half of [hs;h_t]
  }
}

// inp = h_t @ W_inp + b_inp
__global__ __launch_bounds__(256) void inp_kernel(const unsigned short* __restrict__ hsht,
                                                  const unsigned short* __restrict__ Winp,
                                                  const float* __restrict__ b_inp,
                                                  float* __restrict__ inp) {
  const int wid = blockIdx.x * 8 + (threadIdx.x >> 5);
  const int m0  = (wid >> 4) * 16;
  const int n0  = (wid & 15) * 16;
  v8f acc = {};
#pragma unroll
  for (int k = 0; k < Hn; k += 32) {
    v16bf a = load_a_bf16(hsht + Hn, H2, m0, k);  // column offset H, stride 2H
    v16bf b = load_b_packed(Winp, Hn, k, n0);
    acc = wmma_bf16(a, b, acc);
  }
  const int lane = threadIdx.x & 31;
  const int n    = n0 + (lane & 15);
  const int mb   = m0 + ((lane >> 4) << 3);
  const float bias = b_inp[n];
#pragma unroll
  for (int v = 0; v < 8; ++v) inp[(mb + v) * Hn + n] = acc[v] + bias;
}

// Per-batch attention: bf16 score pass, masked softmax, hs, pointer sampling.
__global__ __launch_bounds__(256) void attn_kernel(int s,
                                                   const float* __restrict__ inp,
                                                   const float* __restrict__ ctx,
                                                   const unsigned short* __restrict__ ctx16,
                                                   const float* __restrict__ Vv,
                                                   float* __restrict__ mask,
                                                   const float* __restrict__ emb,
                                                   unsigned short* __restrict__ xh,
                                                   unsigned short* __restrict__ hsht,
                                                   float* __restrict__ out_alpha,
                                                   float* __restrict__ out_ptr) {
  __shared__ float s_inp[Hn];
  __shared__ float s_V[Hn];
  __shared__ float s_att[Ln];
  __shared__ float s_red[256];
  __shared__ int   s_idx;
  const int b = blockIdx.x;
  const int t = threadIdx.x;
  s_inp[t] = inp[b * Hn + t];
  s_V[t]   = Vv[t];
  __syncthreads();

  const float*          ctxb  = ctx   + (size_t)b * Ln * Hn;
  const unsigned short* ctxhb = ctx16 + (size_t)b * Ln * Hn;
  const int wave = t >> 5, lane = t & 31;

  // scores from the bf16 ctx mirror: lanes parallel over K, wave32 xor-reduce
  for (int l = wave * 64; l < wave * 64 + 64; ++l) {
    const int lp = (l + 8 < Ln) ? l + 8 : Ln - 1;
    __builtin_prefetch(ctxhb + (size_t)lp * Hn + lane, 0, 3);  // global_prefetch_b8
    float sum = 0.0f;
#pragma unroll
    for (int kk = 0; kk < Hn; kk += 32) {
      const int k = kk + lane;
      sum += s_V[k] * tanhf(s_inp[k] + bf2f(ctxhb[(size_t)l * Hn + k]));
    }
#pragma unroll
    for (int off = 16; off > 0; off >>= 1) sum += __shfl_xor(sum, off, 32);
    if (lane == 0) s_att[l] = (mask[b * Ln + l] > 0.0f) ? sum : -INFINITY;
  }
  __syncthreads();

  // masked softmax over L=512 (each thread owns 2 slots)
  s_red[t] = fmaxf(s_att[t], s_att[t + 256]);
  __syncthreads();
  for (int off = 128; off > 0; off >>= 1) {
    if (t < off) s_red[t] = fmaxf(s_red[t], s_red[t + off]);
    __syncthreads();
  }
  const float amax = s_red[0];
  __syncthreads();
  const float e0 = __expf(s_att[t] - amax);
  const float e1 = __expf(s_att[t + 256] - amax);
  s_red[t] = e0 + e1;
  __syncthreads();
  for (int off = 128; off > 0; off >>= 1) {
    if (t < off) s_red[t] += s_red[t + off];
    __syncthreads();
  }
  const float inv = 1.0f / (s_red[0] + 1e-20f);
  __syncthreads();
  s_att[t]       = e0 * inv;
  s_att[t + 256] = e1 * inv;
  __syncthreads();

  // emit alpha row for this step
  float* arow = out_alpha + (size_t)b * Ln * Ln + (size_t)s * Ln;
  arow[t]       = s_att[t];
  arow[t + 256] = s_att[t + 256];

  // hs[k] = sum_l alpha[l] * ctx[b,l,k]  (f32 copy, thread t <-> k, coalesced)
  float hs = 0.0f;
  for (int l = 0; l < Ln; ++l) hs += s_att[l] * ctxb[(size_t)l * Hn + t];
  hsht[b * H2 + t] = f2bf(hs);

  // multinomial pointer pick (counter-based hash RNG), mask update
  if (t == 0) {
    unsigned x = (unsigned)(s * 0x9E3779B9u) ^ (unsigned)(b * 0x85EBCA6Bu) ^ 0xC2B2AE35u;
    x ^= x >> 16; x *= 0x7FEB352Du; x ^= x >> 15; x *= 0x846CA68Bu; x ^= x >> 16;
    const float u = (float)(x >> 8) * (1.0f / 16777216.0f);
    float c = 0.0f;
    int idx = -1;
    for (int l = 0; l < Ln; ++l) {
      c += s_att[l];
      if (u < c) { idx = l; break; }
    }
    if (idx < 0) {
      for (int l = Ln - 1; l >= 0; --l)
        if (mask[b * Ln + l] > 0.0f) { idx = l; break; }
      if (idx < 0) idx = 0;
    }
    s_idx = idx;
    mask[b * Ln + idx] = 0.0f;
    out_ptr[b * Ln + s] = (float)idx;
  }
  __syncthreads();
  const int idx = s_idx;
  if (t < En) xh[b * KXH + t] = f2bf(emb[(size_t)b * Ln * En + (size_t)idx * En + t]);
}

// h_new = tanh([hs;h_t] @ W_out + b_out) -> hstate(f32) and xh[:,E:] (bf16)
__global__ __launch_bounds__(256) void hout_kernel(const unsigned short* __restrict__ hsht,
                                                   const unsigned short* __restrict__ Wout,
                                                   const float* __restrict__ b_out,
                                                   float* __restrict__ hstate,
                                                   unsigned short* __restrict__ xh) {
  const int wid = blockIdx.x * 8 + (threadIdx.x >> 5);
  const int m0  = (wid >> 4) * 16;
  const int n0  = (wid & 15) * 16;
  v8f acc = {};
#pragma unroll
  for (int k = 0; k < H2; k += 32) {
    v16bf a = load_a_bf16(hsht, H2, m0, k);
    v16bf b = load_b_packed(Wout, Hn, k, n0);
    acc = wmma_bf16(a, b, acc);
  }
  const int lane = threadIdx.x & 31;
  const int n    = n0 + (lane & 15);
  const int mb   = m0 + ((lane >> 4) << 3);
  const float bias = b_out[n];
#pragma unroll
  for (int v = 0; v < 8; ++v) {
    const int m = mb + v;
    const float hv = tanhf(acc[v] + bias);
    hstate[m * Hn + n]   = hv;
    xh[m * KXH + En + n] = f2bf(hv);
  }
}

__global__ void finalize_kernel(const float* __restrict__ hstate,
                                const float* __restrict__ cstate,
                                float* __restrict__ out_h,
                                float* __restrict__ out_c) {
  __builtin_amdgcn_s_wait_tensorcnt(0);
  int i = blockIdx.x * blockDim.x + threadIdx.x;
  if (i < Bn * Hn) {
    out_h[i] = hstate[i];
    out_c[i] = cstate[i];
  }
}

} // anonymous namespace

extern "C" void kernel_launch(void* const* d_in, const int* in_sizes, int n_in,
                              void* d_out, int out_size, void* d_ws, size_t ws_size,
                              hipStream_t stream) {
  (void)in_sizes; (void)n_in; (void)out_size; (void)ws_size;

  const float* emb   = (const float*)d_in[0];
  const float* dec   = (const float*)d_in[1];
  const float* h0    = (const float*)d_in[2];
  const float* c0    = (const float*)d_in[3];
  const float* ctxin = (const float*)d_in[4];
  const float* W_ih  = (const float*)d_in[5];
  const float* b_ih  = (const float*)d_in[6];
  const float* W_hh  = (const float*)d_in[7];
  const float* b_hh  = (const float*)d_in[8];
  const float* W_out = (const float*)d_in[9];
  const float* b_out = (const float*)d_in[10];
  const float* W_inp = (const float*)d_in[11];
  const float* b_inp = (const float*)d_in[12];
  const float* W_ctx = (const float*)d_in[13];
  const float* b_ctx = (const float*)d_in[14];
  const float* Vv    = (const float*)d_in[15];
  float* out = (float*)d_out;

  // output layout: alpha [B,512,512] | ptrs [B,512] | h_f [B,H] | c_f [B,H]
  float* out_alpha = out;
  float* out_ptr   = out + (size_t)Bn * Ln * Ln;
  float* out_h     = out_ptr + (size_t)Bn * Ln;
  float* out_c     = out_h + (size_t)Bn * Hn;

  // workspace carve-up (256B aligned)
  char*  w   = (char*)d_ws;
  size_t off = 0;
  auto carve = [&](size_t bytes) -> void* {
    void* p = w + off;
    off = (off + bytes + 255) & ~(size_t)255;
    return p;
  };
  unsigned short* Wg     = (unsigned short*)carve((size_t)KXH * G4H * 2); // packed panels
  unsigned short* Winp   = (unsigned short*)carve((size_t)Hn * Hn * 2);
  unsigned short* Wout   = (unsigned short*)carve((size_t)H2 * Hn * 2);
  unsigned short* Wctx   = (unsigned short*)carve((size_t)Hn * Hn * 2);
  unsigned short* xh     = (unsigned short*)carve((size_t)Bn * KXH * 2);  // [x;h] bf16
  unsigned short* hsht   = (unsigned short*)carve((size_t)Bn * H2 * 2);   // [hs;h_t] bf16
  float*          cstate = (float*)carve((size_t)Bn * Hn * 4);
  float*          hstate = (float*)carve((size_t)Bn * Hn * 4);
  float*          inp    = (float*)carve((size_t)Bn * Hn * 4);
  float*          mask   = (float*)carve((size_t)Bn * Ln * 4);
  float*          ctx    = (float*)carve((size_t)Bn * Ln * Hn * 4);       // 64 MB f32
  unsigned short* ctx16  = (unsigned short*)carve((size_t)Bn * Ln * Hn * 2); // 32 MB bf16

  // 1) repack weights into WMMA B-fragment panels (bf16)
  {
    int n1 = En * G4H, n2 = Hn * G4H;
    pack_b_kernel<<<(n1 + 255) / 256, 256, 0, stream>>>(Wg, W_ih, En, G4H, 0);
    pack_b_kernel<<<(n2 + 255) / 256, 256, 0, stream>>>(Wg, W_hh, Hn, G4H, En);
    pack_b_kernel<<<(Hn * Hn + 255) / 256, 256, 0, stream>>>(Winp, W_inp, Hn, Hn, 0);
    pack_b_kernel<<<(H2 * Hn + 255) / 256, 256, 0, stream>>>(Wout, W_out, H2, Hn, 0);
    pack_b_kernel<<<(Hn * Hn + 255) / 256, 256, 0, stream>>>(Wctx, W_ctx, Hn, Hn, 0);
  }

  // 2) per-call state init (deterministic across graph replays)
  init_state_kernel<<<(Bn * Ln) / 256, 256, 0, stream>>>(dec, h0, c0, xh, cstate, mask);

  // 3) one-time context projection via WMMA: 65536x256x256 (f32 + bf16 mirror)
  ctx_proj_kernel<<<(Bn * Ln / 16) * (Hn / 16) / 8, 256, 0, stream>>>(
      ctxin, Wctx, b_ctx, ctx, ctx16);

  // 4) sequential decode: 512 steps x {gates+cell, inp, attention, output}
  for (int s = 0; s < Ln; ++s) {
    lstm_gates_kernel<<<16, 256, 0, stream>>>(xh, Wg, b_ih, b_hh, cstate, hsht);
    inp_kernel<<<16, 256, 0, stream>>>(hsht, Winp, b_inp, inp);
    attn_kernel<<<Bn, 256, 0, stream>>>(s, inp, ctx, ctx16, Vv, mask, emb, xh, hsht,
                                        out_alpha, out_ptr);
    hout_kernel<<<16, 256, 0, stream>>>(hsht, Wout, b_out, hstate, xh);
  }

  // 5) final h/c
  finalize_kernel<<<(Bn * Hn + 255) / 256, 256, 0, stream>>>(hstate, cstate, out_h, out_c);
}
